// Block_28157805592870
// MI455X (gfx1250) — matrix operand order repaired and checked by reading, not compiled, not run on hardware
//
#include <hip/hip_runtime.h>
#include <math.h>

typedef __attribute__((ext_vector_type(16))) _Float16 v16h;
typedef __attribute__((ext_vector_type(8)))  _Float16 v8h;
typedef __attribute__((ext_vector_type(8)))  float    v8f;

#define WMMA_F16(a, b, c) \
  __builtin_amdgcn_wmma_f32_16x16x32_f16(false, (a), false, (b), (short)0, (c), false, false)

constexpr int Bc = 2, Tc = 2048, Cc = 1024, Hc = 16, Dc = 64;
constexpr int C3 = 3 * Cc;
constexpr int MTOT = Bc * Tc;              // 4096 rows
constexpr float ATTN_SCALE = 0.125f;       // 1/sqrt(64)

// ---------------------------------------------------------------------------
// Weight pack: f32 [K,N] row-major  ->  f16 WMMA-B-fragment order.
// Fragment layout (per 32-K chunk kc, per 16-col tile nt):
//   lane L (n = L&15, hi = L>>4), half jj  <->  W[kc*32 + 16*hi + jj][nt*16 + n]
// flat packed index = (((kc * (N/16) + nt) * 32 + L) * 16 + jj)
// Every B-fragment becomes one contiguous, 32B-aligned 16-half load.
// ---------------------------------------------------------------------------
__global__ void pack_w_kernel(const float* __restrict__ W, _Float16* __restrict__ out,
                              int N, size_t total) {
  size_t i = (size_t)blockIdx.x * blockDim.x + threadIdx.x;
  const size_t stride = (size_t)gridDim.x * blockDim.x;
  const int ntiles = N >> 4;
  for (; i < total; i += stride) {
    const int jj = (int)(i & 15);
    const int L  = (int)((i >> 4) & 31);
    const size_t rest = i >> 9;
    const int nt = (int)(rest % ntiles);
    const size_t kc = rest / ntiles;
    const size_t k = kc * 32 + ((L >> 4) << 4) + jj;
    const int col = nt * 16 + (L & 15);
    out[i] = (_Float16)W[k * N + col];
  }
}

// ---------------------------------------------------------------------------
// V pack: from qkv16 [B*T, 3C] -> V in B-fragment order for the P@V WMMAs.
// flat = ((((b*(T/32)+sc)*H + h)*4 + t)*32 + L)*16 + jj
//   <->  V[b, s = sc*32 + 16*hi + jj, h*64 + t*16 + n]
// ---------------------------------------------------------------------------
__global__ void pack_v_kernel(const _Float16* __restrict__ qkv, _Float16* __restrict__ out,
                              size_t total) {
  size_t i = (size_t)blockIdx.x * blockDim.x + threadIdx.x;
  const size_t stride = (size_t)gridDim.x * blockDim.x;
  for (; i < total; i += stride) {
    const int jj = (int)(i & 15);
    const int L  = (int)((i >> 4) & 31);
    const int t  = (int)((i >> 9) & 3);
    const int h  = (int)((i >> 11) & (Hc - 1));
    const size_t bsc = i >> 15;                 // b*(T/32) + sc
    const size_t b = bsc / (Tc / 32);
    const size_t sc = bsc % (Tc / 32);
    const size_t row = b * Tc + sc * 32 + ((L >> 4) << 4) + jj;
    const int col = 2 * Cc + h * Dc + t * 16 + (L & 15);
    out[i] = qkv[row * C3 + col];
  }
}

// ---------------------------------------------------------------------------
// LayerNorm (f32 in, f16 out), one block of 256 threads per row of C=1024
// ---------------------------------------------------------------------------
__global__ __launch_bounds__(256) void ln_f16_kernel(const float* __restrict__ x,
                                                     const float* __restrict__ w,
                                                     const float* __restrict__ bias,
                                                     _Float16* __restrict__ out) {
  __shared__ float red[256];
  const int tid = threadIdx.x;
  const size_t row = blockIdx.x;
  const float4 xv = ((const float4*)(x + row * Cc))[tid];
  red[tid] = xv.x + xv.y + xv.z + xv.w;
  __syncthreads();
  for (int off = 128; off > 0; off >>= 1) {
    if (tid < off) red[tid] += red[tid + off];
    __syncthreads();
  }
  const float mean = red[0] * (1.0f / Cc);
  __syncthreads();
  const float d0 = xv.x - mean, d1 = xv.y - mean, d2 = xv.z - mean, d3 = xv.w - mean;
  red[tid] = d0 * d0 + d1 * d1 + d2 * d2 + d3 * d3;
  __syncthreads();
  for (int off = 128; off > 0; off >>= 1) {
    if (tid < off) red[tid] += red[tid + off];
    __syncthreads();
  }
  const float rstd = rsqrtf(red[0] * (1.0f / Cc) + 1e-5f);
  const int c0 = tid * 4;
  _Float16* o = out + row * Cc + c0;
  o[0] = (_Float16)(d0 * rstd * w[c0 + 0] + bias[c0 + 0]);
  o[1] = (_Float16)(d1 * rstd * w[c0 + 1] + bias[c0 + 1]);
  o[2] = (_Float16)(d2 * rstd * w[c0 + 2] + bias[c0 + 2]);
  o[3] = (_Float16)(d3 * rstd * w[c0 + 3] + bias[c0 + 3]);
}

// ---------------------------------------------------------------------------
// Tiled WMMA GEMM: out[M,N] = A[M,K](f16) @ Wp(packed f16) + bias (+GELU)(+res)
// Block: 128 threads = 4 waves; block tile 64x128; each wave owns 32x64.
// A staged through LDS (vectorized v8h reads); B-fragments loaded directly
// from the fragment-packed weights as single 32B vector loads (L2 resident).
// ---------------------------------------------------------------------------
template <bool OUT_HALF, bool GELU_ACT, bool RESID>
__global__ __launch_bounds__(128) void gemm_wmma_kernel(const _Float16* __restrict__ A,
                                                        const _Float16* __restrict__ Wp,
                                                        const float* __restrict__ bias,
                                                        const float* __restrict__ res,
                                                        void* __restrict__ out,
                                                        int M, int N, int K) {
  __shared__ __align__(32) _Float16 As[64][48];   // padded stride (96B, 32B-mult)
  const int tid  = threadIdx.x;
  const int lane = tid & 31;
  const int n    = lane & 15;
  const int hi   = lane >> 4;
  const int wave = tid >> 5;
  const int waveM = (wave & 1) * 32;
  const int waveN = (wave >> 1) * 64;
  const int rowBase = blockIdx.y * 64;
  const int colBase = blockIdx.x * 128;
  const int ntiles = N >> 4;
  const int ntBase = ((colBase + waveN) >> 4);    // first of 4 col tiles for this wave

  const int ar = tid >> 1, ac = (tid & 1) * 16;   // A tile: 64x32 halves

  v8f acc[2][4] = {};

  for (int k0 = 0; k0 < K; k0 += 32) {
    __syncthreads();
    *(v16h*)&As[ar][ac] = *(const v16h*)&A[(size_t)(rowBase + ar) * K + k0 + ac];
    if (k0 + 32 < K) {
      __builtin_prefetch(&A[(size_t)(rowBase + ar) * K + k0 + 32 + ac], 0, 1);
    }
    __syncthreads();

    v16h afrag[2];
#pragma unroll
    for (int rt = 0; rt < 2; ++rt) {
      const _Float16* arow = &As[waveM + rt * 16 + n][0];
      v8h lo = *(const v8h*)(arow + 16 * hi);
      v8h hh = *(const v8h*)(arow + 16 * hi + 8);
      afrag[rt] = __builtin_shufflevector(lo, hh, 0, 1, 2, 3, 4, 5, 6, 7,
                                          8, 9, 10, 11, 12, 13, 14, 15);
    }
    const size_t kcBase = (size_t)(k0 >> 5) * ntiles;
#pragma unroll
    for (int ct = 0; ct < 4; ++ct) {
      const v16h bfrag =
          *(const v16h*)&Wp[(((kcBase + ntBase + ct) * 32) + lane) * 16];
#pragma unroll
      for (int rt = 0; rt < 2; ++rt) acc[rt][ct] = WMMA_F16(afrag[rt], bfrag, acc[rt][ct]);
    }
  }

#pragma unroll
  for (int rt = 0; rt < 2; ++rt) {
#pragma unroll
    for (int ct = 0; ct < 4; ++ct) {
#pragma unroll
      for (int v = 0; v < 8; ++v) {
        const int row = rowBase + waveM + rt * 16 + v + 8 * hi;
        const int col = colBase + waveN + ct * 16 + n;
        float val = acc[rt][ct][v] + bias[col];
        if constexpr (GELU_ACT) val = 0.5f * val * (1.0f + erff(val * 0.7071067811865475f));
        if constexpr (RESID) val += res[(size_t)row * N + col];
        if constexpr (OUT_HALF)
          ((_Float16*)out)[(size_t)row * N + col] = (_Float16)val;
        else
          ((float*)out)[(size_t)row * N + col] = val;
      }
    }
  }
}

// ---------------------------------------------------------------------------
// Causal attention, one wave per (head, 16-row tile). 3-pass flash style.
// qkv layout: [B*T, 3C] f16 with Q at +0, K at +C, V at +2C, head offset h*D.
// ---------------------------------------------------------------------------
__device__ __forceinline__ v8f score_tile(const _Float16* __restrict__ qkv, int bsel, int j,
                                          int h, v16h aq0, v16h aq1, int n, int hi) {
  // B-frag of K^T: lane n = key position within tile, contiguous along d.
  const _Float16* kp = qkv + ((size_t)(bsel * Tc + j * 16 + n)) * C3 + Cc + h * Dc + 16 * hi;
  v16h b0 = *(const v16h*)kp;          // d = 0..31 slice
  v16h b1 = *(const v16h*)(kp + 32);   // d = 32..63 slice
  v8f acc = {};
  acc = WMMA_F16(aq0, b0, acc);
  acc = WMMA_F16(aq1, b1, acc);
  return acc;
}

__global__ __launch_bounds__(32) void attn_kernel(const _Float16* __restrict__ qkv,
                                                  const _Float16* __restrict__ vpack,
                                                  _Float16* __restrict__ attn_out,
                                                  float* __restrict__ probs, int bsel) {
  __shared__ __align__(32) _Float16 Els[16][32];
  const int lane = threadIdx.x;
  const int n = lane & 15, hi = lane >> 4;
  const int i = blockIdx.x;   // query row tile (16 rows)
  const int h = blockIdx.y;   // head

  // ---- Q A-frags (rows i*16..i*16+15, d split 0..31 / 32..63) ----
  const _Float16* qrow = qkv + ((size_t)(bsel * Tc + i * 16 + n)) * C3 + h * Dc;
  v8h lo = *(const v8h*)(qrow + 16 * hi);
  v8h hh = *(const v8h*)(qrow + 16 * hi + 8);
  const v16h aq0 = __builtin_shufflevector(lo, hh, 0, 1, 2, 3, 4, 5, 6, 7,
                                           8, 9, 10, 11, 12, 13, 14, 15);
  lo = *(const v8h*)(qrow + 32 + 16 * hi);
  hh = *(const v8h*)(qrow + 40 + 16 * hi);
  const v16h aq1 = __builtin_shufflevector(lo, hh, 0, 1, 2, 3, 4, 5, 6, 7,
                                           8, 9, 10, 11, 12, 13, 14, 15);

  // ---- Pass 1: row max ----
  float pm[8];
#pragma unroll
  for (int v = 0; v < 8; ++v) pm[v] = -INFINITY;
  for (int j = 0; j <= i; ++j) {
    v8f s = score_tile(qkv, bsel, j, h, aq0, aq1, n, hi);
#pragma unroll
    for (int v = 0; v < 8; ++v) {
      float sv = s[v] * ATTN_SCALE;
      sv = (j == i && n > v + 8 * hi) ? -INFINITY : sv;
      pm[v] = fmaxf(pm[v], sv);
    }
  }
  float rm[8];
#pragma unroll
  for (int v = 0; v < 8; ++v) {
    float t = pm[v];
    t = fmaxf(t, __shfl_xor(t, 1, 32));
    t = fmaxf(t, __shfl_xor(t, 2, 32));
    t = fmaxf(t, __shfl_xor(t, 4, 32));
    t = fmaxf(t, __shfl_xor(t, 8, 32));
    rm[v] = t;
  }

  // ---- Pass 2: exp-sum + O += E @ V (pairs of 16-col tiles -> K=32) ----
  float rs[8] = {0, 0, 0, 0, 0, 0, 0, 0};
  v8f o[4] = {};
  for (int j0 = 0; j0 <= i; j0 += 2) {
    float e0[8], e1[8];
    {
      v8f s = score_tile(qkv, bsel, j0, h, aq0, aq1, n, hi);
#pragma unroll
      for (int v = 0; v < 8; ++v) {
        float sv = s[v] * ATTN_SCALE;
        sv = (j0 == i && n > v + 8 * hi) ? -INFINITY : sv;
        e0[v] = expf(sv - rm[v]);
      }
    }
    if (j0 + 1 <= i) {
      v8f s = score_tile(qkv, bsel, j0 + 1, h, aq0, aq1, n, hi);
#pragma unroll
      for (int v = 0; v < 8; ++v) {
        float sv = s[v] * ATTN_SCALE;
        sv = (j0 + 1 == i && n > v + 8 * hi) ? -INFINITY : sv;
        e1[v] = expf(sv - rm[v]);
      }
    } else {
#pragma unroll
      for (int v = 0; v < 8; ++v) e1[v] = 0.0f;
    }
#pragma unroll
    for (int v = 0; v < 8; ++v) rs[v] += e0[v] + e1[v];

    __syncthreads();
#pragma unroll
    for (int v = 0; v < 8; ++v) {
      const int m = v + 8 * hi;
      Els[m][n]      = (_Float16)e0[v];
      Els[m][16 + n] = (_Float16)e1[v];
    }
    __syncthreads();

    // A-frag of E (16x32) from LDS, then 4 WMMAs against packed V fragments
    const _Float16* er = &Els[n][0];
    v8h plo = *(const v8h*)(er + 16 * hi);
    v8h phi = *(const v8h*)(er + 16 * hi + 8);
    const v16h aP = __builtin_shufflevector(plo, phi, 0, 1, 2, 3, 4, 5, 6, 7,
                                            8, 9, 10, 11, 12, 13, 14, 15);
    const _Float16* vb =
        vpack + ((size_t)(((bsel * (Tc / 32) + (j0 >> 1)) * Hc + h) * 4) * 32) * 16;
#pragma unroll
    for (int t = 0; t < 4; ++t) {
      const v16h bv = *(const v16h*)(vb + ((size_t)t * 32 + lane) * 16);
      o[t] = WMMA_F16(aP, bv, o[t]);
    }
  }

  float inv[8];
#pragma unroll
  for (int v = 0; v < 8; ++v) {
    float t = rs[v];
    t += __shfl_xor(t, 1, 32);
    t += __shfl_xor(t, 2, 32);
    t += __shfl_xor(t, 4, 32);
    t += __shfl_xor(t, 8, 32);
    inv[v] = 1.0f / t;
  }

  // ---- write attention output tile (16 x 64) as f16 ----
#pragma unroll
  for (int t = 0; t < 4; ++t) {
#pragma unroll
    for (int v = 0; v < 8; ++v) {
      const int m = v + 8 * hi;
      attn_out[((size_t)(bsel * Tc + i * 16 + m)) * Cc + h * Dc + t * 16 + n] =
          (_Float16)(o[t][v] * inv[v]);
    }
  }

  // ---- Pass 3: write 0.5 * probs into d_out mean buffer [H,T,T] ----
  const int jend = (bsel == 0) ? (Tc / 16 - 1) : i;   // b=0 also zero-fills upper triangle
  for (int j = 0; j <= jend; ++j) {
    float p[8];
    if (j <= i) {
      v8f s = score_tile(qkv, bsel, j, h, aq0, aq1, n, hi);
#pragma unroll
      for (int v = 0; v < 8; ++v) {
        float sv = s[v] * ATTN_SCALE;
        sv = (j == i && n > v + 8 * hi) ? -INFINITY : sv;
        p[v] = expf(sv - rm[v]) * inv[v];
      }
    } else {
#pragma unroll
      for (int v = 0; v < 8; ++v) p[v] = 0.0f;
    }
#pragma unroll
    for (int v = 0; v < 8; ++v) {
      const int m = v + 8 * hi;
      float* addr = probs + (size_t)h * Tc * Tc + (size_t)(i * 16 + m) * Tc + j * 16 + n;
      if (bsel == 0) *addr = 0.5f * p[v];
      else           *addr += 0.5f * p[v];
    }
  }
}

// ---------------------------------------------------------------------------
// Host-side orchestration
// ---------------------------------------------------------------------------
extern "C" void kernel_launch(void* const* d_in, const int* in_sizes, int n_in,
                              void* d_out, int out_size, void* d_ws, size_t ws_size,
                              hipStream_t stream) {
  const float* x      = (const float*)d_in[0];
  const float* ln1_w  = (const float*)d_in[1];
  const float* ln1_b  = (const float*)d_in[2];
  const float* w_qkv  = (const float*)d_in[3];
  const float* b_qkv  = (const float*)d_in[4];
  const float* w_proj = (const float*)d_in[5];
  const float* b_proj = (const float*)d_in[6];
  const float* ln2_w  = (const float*)d_in[7];
  const float* ln2_b  = (const float*)d_in[8];
  const float* w_fc   = (const float*)d_in[9];
  const float* b_fc   = (const float*)d_in[10];
  const float* w_fc2  = (const float*)d_in[11];
  const float* b_fc2  = (const float*)d_in[12];

  float* out_x   = (float*)d_out;
  float* out_att = (float*)d_out + (size_t)MTOT * Cc;

  char* ws = (char*)d_ws;
  size_t off = 0;
  auto carve = [&](size_t bytes) { char* p = ws + off; off += (bytes + 255) & ~(size_t)255; return p; };
  _Float16* wqkv16  = (_Float16*)carve((size_t)Cc * 3 * Cc * 2);
  _Float16* wproj16 = (_Float16*)carve((size_t)Cc * Cc * 2);
  _Float16* wfc16   = (_Float16*)carve((size_t)Cc * 4 * Cc * 2);
  _Float16* wfc216  = (_Float16*)carve((size_t)4 * Cc * Cc * 2);
  _Float16* h16     = (_Float16*)carve((size_t)MTOT * Cc * 2);
  _Float16* qkv16   = (_Float16*)carve((size_t)MTOT * 3 * Cc * 2);
  _Float16* vpack   = (_Float16*)carve((size_t)MTOT * Cc * 2);
  _Float16* attn16  = (_Float16*)carve((size_t)MTOT * Cc * 2);
  float*    x1      = (float*)carve((size_t)MTOT * Cc * 4);
  _Float16* h2_16   = (_Float16*)carve((size_t)MTOT * Cc * 2);
  _Float16* hfc16   = (_Float16*)carve((size_t)MTOT * 4 * Cc * 2);

  // 1) Weight conversion + fragment packing (f32 -> f16, B-frag order)
  pack_w_kernel<<<2048, 256, 0, stream>>>(w_qkv,  wqkv16,  3 * Cc, (size_t)Cc * 3 * Cc);
  pack_w_kernel<<<2048, 256, 0, stream>>>(w_proj, wproj16, Cc,     (size_t)Cc * Cc);
  pack_w_kernel<<<2048, 256, 0, stream>>>(w_fc,   wfc16,   4 * Cc, (size_t)Cc * 4 * Cc);
  pack_w_kernel<<<2048, 256, 0, stream>>>(w_fc2,  wfc216,  Cc,     (size_t)4 * Cc * Cc);

  // 2) LN1
  ln_f16_kernel<<<MTOT, 256, 0, stream>>>(x, ln1_w, ln1_b, h16);

  // 3) QKV GEMM: [4096,1024] x [1024,3072] -> f16
  gemm_wmma_kernel<true, false, false><<<dim3(3 * Cc / 128, MTOT / 64), 128, 0, stream>>>(
      h16, wqkv16, b_qkv, nullptr, qkv16, MTOT, 3 * Cc, Cc);

  // 4) Pack V into P@V fragment order
  pack_v_kernel<<<2048, 256, 0, stream>>>(qkv16, vpack, (size_t)MTOT * Cc);

  // 5) Attention (two stream-ordered launches so b=1 can accumulate into probs)
  attn_kernel<<<dim3(Tc / 16, Hc), 32, 0, stream>>>(qkv16, vpack, attn16, out_att, 0);
  attn_kernel<<<dim3(Tc / 16, Hc), 32, 0, stream>>>(qkv16, vpack, attn16, out_att, 1);

  // 6) Projection + residual: x1 = x + attn @ w_proj + b_proj (f32)
  gemm_wmma_kernel<false, false, true><<<dim3(Cc / 128, MTOT / 64), 128, 0, stream>>>(
      attn16, wproj16, b_proj, x, x1, MTOT, Cc, Cc);

  // 7) LN2
  ln_f16_kernel<<<MTOT, 256, 0, stream>>>(x1, ln2_w, ln2_b, h2_16);

  // 8) FC + exact GELU: [4096,1024] x [1024,4096] -> f16
  gemm_wmma_kernel<true, true, false><<<dim3(4 * Cc / 128, MTOT / 64), 128, 0, stream>>>(
      h2_16, wfc16, b_fc, nullptr, hfc16, MTOT, 4 * Cc, Cc);

  // 9) FC2 + residual -> d_out x region (f32)
  gemm_wmma_kernel<false, false, true><<<dim3(Cc / 128, MTOT / 64), 128, 0, stream>>>(
      hfc16, wfc216, b_fc2, x1, out_x, MTOT, Cc, 4 * Cc);
}
